// PerModalityStateTokenizer_82471962017993
// MI455X (gfx1250) — matrix operand order repaired and checked by reading, not compiled, not run on hardware
//
#include <hip/hip_runtime.h>

// ---------------------------------------------------------------------------
// PerModalityStateTokenizer for MI455X (gfx1250, wave32, WMMA).
//
// Strategy: memory-bound op (AI ~ 8 FLOP/B). Group the 512 tokens by
// embodiment id, run 16-row dense GEMM tiles with v_wmma_f32_16x16x32_f16,
// fused layer1(relu) -> LDS -> layer2 so W2 (268 MB total) is streamed
// once per tile (L2 is 192 MB, catches most cross-tile reuse).
// ---------------------------------------------------------------------------

typedef __attribute__((ext_vector_type(16))) _Float16 v16h;
typedef __attribute__((ext_vector_type(8)))  float    v8f;

#define HID   512
#define EMB   1024
#define NMOD  4
#define NTOK  512          // B*T = 16*32
#define DSTATE 192         // 64+64+32+32
#define MAXTILES 64        // worst case sum(ceil(cnt_e/16)) = 62

// Workspace (int) layout: [0] = ntiles, [1..512] = sorted token ids,
// [520 + 3*t ..] = tile descriptors {emb, start, rows}.
#define WS_SORT 1
#define WS_DESC 520

struct Params {
  const float* W1[4]; const float* b1[4];
  const float* W2[4]; const float* b2[4];
  const float* te[4];
};

// ---------------------------------------------------------------------------
// Kernel 1: bucket tokens by embodiment id, emit padded 16-row tile descs.
// Single 512-thread block; trivial cost.
// ---------------------------------------------------------------------------
__global__ __launch_bounds__(512)
void group_tokens(const int* __restrict__ ids, int* __restrict__ ws) {
  __shared__ int cnt[32];
  __shared__ int off[32];
  const int tid = threadIdx.x;
  if (tid < 32) cnt[tid] = 0;
  __syncthreads();
  const int c = ids[tid] & 31;                 // embodiment id of this token
  const int pos = atomicAdd(&cnt[c], 1);       // rank within bucket (order-free)
  __syncthreads();
  if (tid == 0) {
    int o = 0, nt = 0;
    for (int e = 0; e < 32; ++e) { off[e] = o; o += cnt[e]; }
    for (int e = 0; e < 32; ++e) {
      const int cc = cnt[e];
      for (int k = 0; k < cc; k += 16) {
        ws[WS_DESC + nt * 3 + 0] = e;
        ws[WS_DESC + nt * 3 + 1] = off[e] + k;
        ws[WS_DESC + nt * 3 + 2] = (cc - k < 16) ? (cc - k) : 16;
        ++nt;
      }
    }
    ws[0] = nt;
  }
  __syncthreads();
  ws[WS_SORT + off[c] + pos] = tid;
}

// ---------------------------------------------------------------------------
// Kernel 2: fused 2-layer MLP per (tile, modality) via WMMA.
// 512 threads = 16 wave32s. All WMMAs execute with EXEC all-ones
// (only uniform early-exit; per-lane divergence only at final stores).
// ---------------------------------------------------------------------------
__global__ __launch_bounds__(512)
void tokenize_fused(const float* __restrict__ state,
                    const int* __restrict__ ws,
                    Params p,
                    float* __restrict__ out) {
  const int tile = blockIdx.x;
  const int m    = blockIdx.y;
  if (tile >= ws[0]) return;                       // uniform per block

  const int e    = ws[WS_DESC + tile * 3 + 0];
  const int strt = ws[WS_DESC + tile * 3 + 1];
  const int rows = ws[WS_DESC + tile * 3 + 2];

  const int L   = (m < 2) ? 64 : 32;               // modality slice length
  const int off = (m < 2) ? (m * 64) : (128 + (m - 2) * 32);

  // LDS: A-operand staging. Row strides padded to avoid 16-way bank conflicts
  // on the A-fragment gathers (all 16 low lanes read the same K column).
  __shared__ _Float16 xA[16][64 + 8];              // layer-1 A tile (f16)
  __shared__ _Float16 hX[16][HID + 8];             // layer-1 output / layer-2 A
  __shared__ int toks[16];

  const int tid  = threadIdx.x;
  const int lane = tid & 31;
  const int wave = tid >> 5;
  const int hi16 = (lane >> 4) & 1;                // upper half-wave flag
  const int ln   = lane & 15;

  if (tid < 16) toks[tid] = (tid < rows) ? ws[WS_SORT + strt + tid] : 0;
  // zero the A tile (padded rows / padded K contribute 0 to the GEMM)
  for (int i = tid; i < 16 * (64 + 8); i += 512)
    xA[i / (64 + 8)][i % (64 + 8)] = (_Float16)0.f;
  __syncthreads();
  for (int i = tid; i < rows * L; i += 512) {
    const int r = i / L, k = i % L;
    xA[r][k] = (_Float16)state[(size_t)toks[r] * DSTATE + off + k];
  }
  __syncthreads();

  // ---------------- Layer 1: h = relu(x @ W1[e] + b1[e]) -------------------
  // W1[m]: (32, L, HID) row-major -> B matrix is (K=L) x (N=HID).
  const float* W1 = p.W1[m] + (size_t)e * L * HID;
  const float* B1 = p.b1[m] + (size_t)e * HID;
  #pragma unroll
  for (int j = 0; j < 2; ++j) {
    const int nt = wave + 16 * j;                  // 0..31 N-tiles of HID
    const int n  = nt * 16 + ln;
    v8f acc = {};
    for (int k0 = 0; k0 < L; k0 += 32) {
      v16h a, b;
      #pragma unroll
      for (int h = 0; h < 16; ++h) {
        // A (16x32 f16) layout: M = lane&15; K striped per ISA 7.12.2
        const int v  = h >> 1, po = h & 1;
        const int ka = k0 + ((v & 4) ? 16 : 0) + (hi16 ? 8 : 0) + ((v & 3) << 1) + po;
        a[h] = xA[ln][ka];
        // B (32x16 f16) layout: N = lane&15; K = ((h&8)?16:0) + (hi16?8:0) + (h&7)
        const int kb = k0 + ((h & 8) ? 16 : 0) + (hi16 ? 8 : 0) + (h & 7);
        b[h] = (_Float16)W1[(size_t)kb * HID + n];
      }
      acc = __builtin_amdgcn_wmma_f32_16x16x32_f16(
          /*neg_a=*/false, a, /*neg_b=*/false, b,
          /*c_mod=*/(short)0, acc, /*reuse_a=*/false, /*reuse_b=*/false);
    }
    const float bias = B1[n];
    #pragma unroll
    for (int r = 0; r < 8; ++r) {                  // C/D: M = r + (hi16?8:0)
      const int M = r + (hi16 ? 8 : 0);
      const float v = acc[r] + bias;
      hX[M][n] = (_Float16)(v > 0.f ? v : 0.f);
    }
  }
  __syncthreads();

  // ---------------- Layer 2: z = h @ W2[e] + b2[e] + te --------------------
  // W2[m]: (32, HID, EMB) row-major -> B matrix is (K=HID) x (N=EMB).
  const float* W2 = p.W2[m] + (size_t)e * HID * EMB;
  const float* B2 = p.b2[m] + (size_t)e * EMB;
  const float* TE = p.te[m];
  #pragma unroll
  for (int j = 0; j < 4; ++j) {
    const int nt = wave + 16 * j;                  // 0..63 N-tiles of EMB
    const int n  = nt * 16 + ln;
    v8f acc = {};
    for (int k0 = 0; k0 < HID; k0 += 32) {
      v16h a, b;
      #pragma unroll
      for (int h = 0; h < 16; ++h) {
        const int v  = h >> 1, po = h & 1;
        const int ka = k0 + ((v & 4) ? 16 : 0) + (hi16 ? 8 : 0) + ((v & 3) << 1) + po;
        a[h] = hX[ln][ka];
        const int kb = k0 + ((h & 8) ? 16 : 0) + (hi16 ? 8 : 0) + (h & 7);
        b[h] = (_Float16)W2[(size_t)kb * EMB + n];
      }
      if (k0 + 32 < HID)                           // gfx1250 global_prefetch_b8
        __builtin_prefetch(&W2[(size_t)(k0 + 32 + (hi16 ? 16 : 0)) * EMB + n], 0, 1);
      acc = __builtin_amdgcn_wmma_f32_16x16x32_f16(
          false, a, false, b, (short)0, acc, false, false);
    }
    const float bias = B2[n] + TE[n];
    #pragma unroll
    for (int r = 0; r < 8; ++r) {
      const int M = r + (hi16 ? 8 : 0);
      if (M < rows) {                              // mask padded rows
        const int tok = toks[M];                   // flat b*T+t index
        out[((size_t)tok * NMOD + m) * EMB + n] = acc[r] + bias;
      }
    }
  }
}

// ---------------------------------------------------------------------------
extern "C" void kernel_launch(void* const* d_in, const int* in_sizes, int n_in,
                              void* d_out, int out_size, void* d_ws, size_t ws_size,
                              hipStream_t stream) {
  (void)in_sizes; (void)n_in; (void)out_size; (void)ws_size;
  const float* state = (const float*)d_in[0];
  const int*   ids   = (const int*)d_in[1];
  Params p;
  for (int m = 0; m < 4; ++m) {
    p.W1[m] = (const float*)d_in[2 + 5 * m + 0];
    p.b1[m] = (const float*)d_in[2 + 5 * m + 1];
    p.W2[m] = (const float*)d_in[2 + 5 * m + 2];
    p.b2[m] = (const float*)d_in[2 + 5 * m + 3];
    p.te[m] = (const float*)d_in[2 + 5 * m + 4];
  }
  int* ws = (int*)d_ws;

  hipLaunchKernelGGL(group_tokens, dim3(1), dim3(512), 0, stream, ids, ws);
  hipLaunchKernelGGL(tokenize_fused, dim3(MAXTILES, NMOD), dim3(512), 0, stream,
                     state, ws, p, (float*)d_out);
}
